// MappingNetwork_6390911337097
// MI455X (gfx1250) — matrix operand order repaired and checked by reading, compiled, not run on hardware
//
#include <hip/hip_runtime.h>

typedef _Float16 f16;
typedef __attribute__((ext_vector_type(16))) _Float16 v16h;
typedef __attribute__((ext_vector_type(8)))  _Float16 v8h;
typedef __attribute__((ext_vector_type(8)))  float    v8f;

#define BATCH 8192
#define LDIM  16
#define HDIM  512
#define SDIM  64
#define NDOM  16
#define ROWS  64            // rows per block tile
#define TPB   128           // 4 waves (wave32)
#define MAXTILES (BATCH / ROWS)   // 128 per domain worst case

__device__ __forceinline__ v16h cat16(v8h lo, v8h hi) {
  return __builtin_shufflevector(lo, hi, 0,1,2,3,4,5,6,7,8,9,10,11,12,13,14,15);
}

// Full layer: [ROWS,512] (LDS) x Wt[512][512] (global, col-major f16) -> [ROWS,512] (LDS)
// with bias + LeakyReLU(0.2), output converted to f16.
// Register blocking: each wave owns 8 of the 32 column tiles (4 pairs) and keeps
// accumulators for all 4 row tiles x 2 column tiles, so each B k-chunk is loaded
// once per pair and each A k-chunk once per row tile => 8 WMMA per 12 b128 loads.
__device__ __forceinline__ void gemm_layer_lds(const f16* __restrict__ in,
                                               const f16* __restrict__ wt,
                                               const float* __restrict__ bias,
                                               f16* __restrict__ out) {
  const int tid = threadIdx.x, wave = tid >> 5, lane = tid & 31;
  const int g = lane >> 4, mr = lane & 15;
  const f16* aBase = in + mr * HDIM + g * 8;     // A: lane's row, k-chunks g*8 / 16+g*8
  for (int cp = 0; cp < 4; ++cp) {
    const int ct0 = wave * 8 + cp * 2;
    const f16* bCol0 = wt + (ct0 * 16 + mr) * HDIM + g * 8;  // B: lane's column, k-chunks g*8 / 16+g*8
    const f16* bCol1 = bCol0 + 16 * HDIM;
    v8f acc[4][2];
#pragma unroll
    for (int rt = 0; rt < 4; ++rt) { acc[rt][0] = (v8f){}; acc[rt][1] = (v8f){}; }
#pragma unroll 2
    for (int k0 = 0; k0 < HDIM; k0 += 32) {
      const v16h B0 = cat16(*(const v8h*)(bCol0 + k0), *(const v8h*)(bCol0 + k0 + 16));
      const v16h B1 = cat16(*(const v8h*)(bCol1 + k0), *(const v8h*)(bCol1 + k0 + 16));
#pragma unroll
      for (int rt = 0; rt < 4; ++rt) {
        const f16* aRow = aBase + rt * 16 * HDIM + k0;
        const v16h A = cat16(*(const v8h*)(aRow), *(const v8h*)(aRow + 16));
        acc[rt][0] = __builtin_amdgcn_wmma_f32_16x16x32_f16(false, A, false, B0,
                                                            (short)0, acc[rt][0], false, false);
        acc[rt][1] = __builtin_amdgcn_wmma_f32_16x16x32_f16(false, A, false, B1,
                                                            (short)0, acc[rt][1], false, false);
      }
    }
#pragma unroll
    for (int cc = 0; cc < 2; ++cc) {
      const int n = (ct0 + cc) * 16 + mr;
      const float bn = bias[n];
#pragma unroll
      for (int rt = 0; rt < 4; ++rt) {
#pragma unroll
        for (int r = 0; r < 8; ++r) {
          float v = acc[rt][cc][r] + bn;
          v = (v >= 0.f) ? v : 0.2f * v;
          out[(rt * 16 + r + 8 * g) * HDIM + n] = (f16)v;
        }
      }
    }
  }
}

// ---------------- weight prep: f32 [K][N] -> f16 [N][K] (per batch slice) -----
__global__ void transpose_convert(const float* __restrict__ in, f16* __restrict__ out,
                                  int K, int N) {
  const long base_in  = (long)blockIdx.y * K * N;
  const long base_out = (long)blockIdx.y * N * K;
  const int idx = blockIdx.x * blockDim.x + threadIdx.x;
  if (idx < K * N) {
    const int k = idx / N, n = idx - k * N;
    out[base_out + (long)n * K + k] = (f16)in[base_in + idx];
  }
}

// ---------------- domain bucketing ----------------
__global__ void zero_cnt(int* cnt) { if (threadIdx.x < NDOM) cnt[threadIdx.x] = 0; }

__global__ void bucket_kernel(const int* __restrict__ y, int* __restrict__ cnt,
                              int* __restrict__ perm) {
  const int b = blockIdx.x * blockDim.x + threadIdx.x;
  if (b < BATCH) {
    const int d = y[b];
    const int pos = atomicAdd(&cnt[d], 1);
    perm[d * BATCH + pos] = b;
  }
}

// ---------------- fused trunk: 4 layers, activations resident in LDS ----------
__global__ __launch_bounds__(TPB) void trunk_kernel(
    const float* __restrict__ z, const float* __restrict__ tW0, const float* __restrict__ tb0,
    const f16* __restrict__ wt1, const float* __restrict__ tb1,
    const f16* __restrict__ wt2, const float* __restrict__ tb2,
    const f16* __restrict__ wt3, const float* __restrict__ tb3,
    f16* __restrict__ common) {
  __shared__ f16 bufA[ROWS * HDIM];
  __shared__ f16 bufB[ROWS * HDIM];
  __shared__ float zt[ROWS * LDIM];
  const int tid = threadIdx.x;
  const int rstart = blockIdx.x * ROWS;

  for (int i = tid; i < ROWS * LDIM; i += TPB) zt[i] = z[rstart * LDIM + i];
  __syncthreads();

  // layer 0: K=16, plain VALU FMA (tiny fraction of total work)
  for (int i = tid; i < ROWS * HDIM; i += TPB) {
    const int r = i >> 9, j = i & (HDIM - 1);
    float acc = tb0[j];
#pragma unroll
    for (int k = 0; k < LDIM; ++k) acc += zt[r * LDIM + k] * tW0[k * HDIM + j];
    bufA[i] = (f16)(acc >= 0.f ? acc : 0.2f * acc);
  }
  __syncthreads();

  gemm_layer_lds(bufA, wt1, tb1, bufB);
  __syncthreads();
  gemm_layer_lds(bufB, wt2, tb2, bufA);
  __syncthreads();
  gemm_layer_lds(bufA, wt3, tb3, bufB);
  __syncthreads();

  // coalesced write of common (f16) to global
  const v8h* src = (const v8h*)bufB;
  v8h* dst = (v8h*)(common + (long)rstart * HDIM);
  for (int i = tid; i < ROWS * HDIM / 8; i += TPB) dst[i] = src[i];
}

// ---------------- fused per-domain heads (routed grouped GEMM) ----------------
__global__ __launch_bounds__(TPB) void heads_kernel(
    const f16* __restrict__ common, const int* __restrict__ cnt, const int* __restrict__ perm,
    const f16* __restrict__ hwt0, const float* __restrict__ hb0,
    const f16* __restrict__ hwt1, const float* __restrict__ hb1,
    const f16* __restrict__ hwt2, const float* __restrict__ hb2,
    const f16* __restrict__ hwt3, const float* __restrict__ hb3,
    float* __restrict__ out) {
  __shared__ f16 bufA[ROWS * HDIM];
  __shared__ f16 bufB[ROWS * HDIM];
  __shared__ int rowIdx[ROWS];

  const int d = blockIdx.x / MAXTILES;
  const int t = blockIdx.x - d * MAXTILES;
  const int c = cnt[d];
  if (t * ROWS >= c) return;  // block-uniform exit (EXEC stays all-ones for WMMA)

  const int tid = threadIdx.x;
  if (tid < ROWS) {
    const int gi = t * ROWS + tid;
    rowIdx[tid] = (gi < c) ? perm[d * BATCH + gi] : -1;
  }
  __syncthreads();

  // gather this bucket tile's `common` rows into LDS (16B-vector copies)
  for (int i = tid; i < ROWS * (HDIM / 8); i += TPB) {
    const int r = i >> 6, off = i & 63;
    int b = rowIdx[r];
    if (b < 0) b = rowIdx[0];  // pad rows duplicate a valid row; never stored
    ((v8h*)bufA)[r * (HDIM / 8) + off] = ((const v8h*)(common + (long)b * HDIM))[off];
  }
  __syncthreads();

  gemm_layer_lds(bufA, hwt0 + (long)d * HDIM * HDIM, hb0 + d * HDIM, bufB);
  __syncthreads();
  gemm_layer_lds(bufB, hwt1 + (long)d * HDIM * HDIM, hb1 + d * HDIM, bufA);
  __syncthreads();
  gemm_layer_lds(bufA, hwt2 + (long)d * HDIM * HDIM, hb2 + d * HDIM, bufB);
  __syncthreads();

  // final layer: [ROWS,512] x Wt3[d][64][512] -> out[b][0..63] (f32, no lrelu)
  // 4 column tiles (S=64) -> exactly one per wave; all 4 row tiles in registers.
  const int wave = tid >> 5, lane = tid & 31, g = lane >> 4, mr = lane & 15;
  const f16* wt3d = hwt3 + (long)d * SDIM * HDIM;
  const float* b3 = hb3 + d * SDIM;
  const int ct = wave;
  const f16* bCol = wt3d + (ct * 16 + mr) * HDIM + g * 8;
  const f16* aBase = bufB + mr * HDIM + g * 8;
  v8f acc[4];
#pragma unroll
  for (int rt = 0; rt < 4; ++rt) acc[rt] = (v8f){};
#pragma unroll 2
  for (int k0 = 0; k0 < HDIM; k0 += 32) {
    const v16h B = cat16(*(const v8h*)(bCol + k0), *(const v8h*)(bCol + k0 + 16));
#pragma unroll
    for (int rt = 0; rt < 4; ++rt) {
      const f16* aRow = aBase + rt * 16 * HDIM + k0;
      const v16h A = cat16(*(const v8h*)(aRow), *(const v8h*)(aRow + 16));
      acc[rt] = __builtin_amdgcn_wmma_f32_16x16x32_f16(false, A, false, B,
                                                       (short)0, acc[rt], false, false);
    }
  }
  const int n = ct * 16 + mr;
  const float bn = b3[n];
#pragma unroll
  for (int rt = 0; rt < 4; ++rt) {
#pragma unroll
    for (int r = 0; r < 8; ++r) {
      const int m = rt * 16 + r + 8 * g;
      const int b = rowIdx[m];
      if (b >= 0) out[(long)b * SDIM + n] = acc[rt][r] + bn;
    }
  }
}

extern "C" void kernel_launch(void* const* d_in, const int* in_sizes, int n_in,
                              void* d_out, int out_size, void* d_ws, size_t ws_size,
                              hipStream_t stream) {
  const float* z   = (const float*)d_in[0];
  const int*   y   = (const int*)d_in[1];
  const float* tW0 = (const float*)d_in[2];
  const float* tb0 = (const float*)d_in[3];
  const float* tW1 = (const float*)d_in[4];
  const float* tb1 = (const float*)d_in[5];
  const float* tW2 = (const float*)d_in[6];
  const float* tb2 = (const float*)d_in[7];
  const float* tW3 = (const float*)d_in[8];
  const float* tb3 = (const float*)d_in[9];
  const float* hW0 = (const float*)d_in[10];
  const float* hb0 = (const float*)d_in[11];
  const float* hW1 = (const float*)d_in[12];
  const float* hb1 = (const float*)d_in[13];
  const float* hW2 = (const float*)d_in[14];
  const float* hb2 = (const float*)d_in[15];
  const float* hW3 = (const float*)d_in[16];
  const float* hb3 = (const float*)d_in[17];
  float* out = (float*)d_out;

  char* p = (char*)d_ws;
  auto take = [&](size_t bytes) { char* r = p; p += (bytes + 255) & ~(size_t)255; return r; };
  f16* wt1    = (f16*)take((size_t)HDIM * HDIM * 2);
  f16* wt2    = (f16*)take((size_t)HDIM * HDIM * 2);
  f16* wt3    = (f16*)take((size_t)HDIM * HDIM * 2);
  f16* hwt0   = (f16*)take((size_t)NDOM * HDIM * HDIM * 2);
  f16* hwt1   = (f16*)take((size_t)NDOM * HDIM * HDIM * 2);
  f16* hwt2   = (f16*)take((size_t)NDOM * HDIM * HDIM * 2);
  f16* hwt3   = (f16*)take((size_t)NDOM * SDIM * HDIM * 2);
  f16* common = (f16*)take((size_t)BATCH * HDIM * 2);
  int* cnt    = (int*)take(NDOM * 4);
  int* perm   = (int*)take((size_t)NDOM * BATCH * 4);

  // weight conversion f32 row-major -> f16 col-major
  const int nHH = HDIM * HDIM;                      // 262144
  transpose_convert<<<dim3((nHH + 255) / 256, 1), 256, 0, stream>>>(tW1, wt1, HDIM, HDIM);
  transpose_convert<<<dim3((nHH + 255) / 256, 1), 256, 0, stream>>>(tW2, wt2, HDIM, HDIM);
  transpose_convert<<<dim3((nHH + 255) / 256, 1), 256, 0, stream>>>(tW3, wt3, HDIM, HDIM);
  transpose_convert<<<dim3((nHH + 255) / 256, NDOM), 256, 0, stream>>>(hW0, hwt0, HDIM, HDIM);
  transpose_convert<<<dim3((nHH + 255) / 256, NDOM), 256, 0, stream>>>(hW1, hwt1, HDIM, HDIM);
  transpose_convert<<<dim3((nHH + 255) / 256, NDOM), 256, 0, stream>>>(hW2, hwt2, HDIM, HDIM);
  const int nHS = HDIM * SDIM;                      // 32768
  transpose_convert<<<dim3((nHS + 255) / 256, NDOM), 256, 0, stream>>>(hW3, hwt3, HDIM, SDIM);

  // domain bucketing
  zero_cnt<<<1, 32, 0, stream>>>(cnt);
  bucket_kernel<<<(BATCH + 255) / 256, 256, 0, stream>>>(y, cnt, perm);

  // fused trunk
  trunk_kernel<<<BATCH / ROWS, TPB, 0, stream>>>(z, tW0, tb0, wt1, tb1, wt2, tb2, wt3, tb3, common);

  // fused routed heads
  heads_kernel<<<NDOM * MAXTILES, TPB, 0, stream>>>(common, cnt, perm,
                                                    hwt0, hb0, hwt1, hb1, hwt2, hb2, hwt3, hb3,
                                                    out);
}
